// CondEmbedRNN_6914897347354
// MI455X (gfx1250) — compile-verified, matrix operand order checked by reading, not run on hardware
//
#include <hip/hip_runtime.h>
#include <hip/hip_bf16.h>

typedef __bf16 bf16;
typedef __attribute__((ext_vector_type(16))) __bf16  v16bf;
typedef __attribute__((ext_vector_type(8)))  __bf16  v8bf;
typedef __attribute__((ext_vector_type(8)))  float   v8f;
typedef __attribute__((ext_vector_type(4)))  unsigned u32x4;
typedef __attribute__((ext_vector_type(8)))  unsigned u32x8;

static constexpr int Bb = 128, Tt = 512, Hh = 1024, IND = 1024, EMD = 1024;
static constexpr int KS = 128;   // K-slice staged in LDS per TDM transfer

// ---------- WMMA helpers ----------------------------------------------------
// 16-bit A/B operand tile (16 rows x 32 K) per the CDNA5 VGPR layout:
// lanes 0-15: row=lane,   K chunks {0..7, 16..23}
// lanes 16-31: row=lane-16, K chunks {8..15, 24..31}
__device__ __forceinline__ v16bf load_tile(const bf16* __restrict__ p, size_t ld, int lane) {
  const int r    = lane & 15;
  const int koff = (lane & 16) ? 8 : 0;
  const bf16* q  = p + (size_t)r * ld + koff;
  v8bf lo = *(const v8bf*)(q);
  v8bf hi = *(const v8bf*)(q + 16);
  v16bf out;
#pragma unroll
  for (int i = 0; i < 8; ++i) { out[i] = lo[i]; out[i + 8] = hi[i]; }
  return out;
}

__device__ __forceinline__ v8f wmma_bf16(v16bf a, v16bf b, v8f c) {
  return __builtin_amdgcn_wmma_f32_16x16x32_bf16(
      /*neg_a=*/false, a, /*neg_b=*/false, b,
      /*c_mod=*/(short)0, c, /*reuse_a=*/false, /*reuse_b=*/false);
}

__device__ __forceinline__ float sigmoidf_(float x) { return 1.0f / (1.0f + __expf(-x)); }

// ---------- Tensor Data Mover (TDM) staging ---------------------------------
// 32-bit LDS byte offset of a __shared__ object (generic -> LDS addrspace cast)
__device__ __forceinline__ unsigned lds_addr32(const void* p) {
  return (unsigned)(uintptr_t)(const __attribute__((address_space(3))) char*)p;
}

// Issue a 2-D TDM load: tile_h rows x tile_w elements of bf16 from a row-major
// matrix with row pitch `row_stride` (elements), compacted contiguously into
// LDS at byte offset lds_off. Descriptor per CDNA5 ISA.
__device__ __forceinline__ void tdm_load_2d(const bf16* gsrc, unsigned lds_off,
                                            unsigned tile_w, unsigned tile_h,
                                            unsigned row_stride) {
  unsigned long long ga = (unsigned long long)(uintptr_t)gsrc;
  u32x4 g0;
  g0[0] = 1u;                                              // count=1, user D#
  g0[1] = lds_off;                                         // lds_addr (bytes)
  g0[2] = (unsigned)ga;                                    // global_addr[31:0]
  g0[3] = (unsigned)((ga >> 32) & 0x1FFFFFFu) | (2u << 30);// addr[56:32] | type=2
  u32x8 g1;
  g1[0] = (1u << 16);                                      // data_size=1 (2B), no mcast
  g1[1] = (tile_w & 0xFFFFu) << 16;                        // tensor_dim0[15:0]
  g1[2] = (tile_w >> 16) | ((tile_h & 0xFFFFu) << 16);     // td0[31:16] | td1[15:0]
  g1[3] = (tile_h >> 16) | (tile_w << 16);                 // td1[31:16] | tile_dim0
  g1[4] = tile_h & 0xFFFFu;                                // tile_dim1 | tile_dim2=0
  g1[5] = row_stride;                                      // tensor_dim0_stride[31:0]
  g1[6] = 0u;                                              // stride hi | dim1_stride lo
  g1[7] = 0u;
  asm volatile("tensor_load_to_lds %0, %1" :: "s"(g0), "s"(g1) : "memory");
}

// ---------- prep kernels ----------------------------------------------------
// out[N x K] bf16 = transpose of in[K x N] fp32
__global__ void k_transpose_bf16(const float* __restrict__ in, bf16* __restrict__ out,
                                 int K, int N) {
  size_t idx = (size_t)blockIdx.x * 256 + threadIdx.x;
  if (idx >= (size_t)N * K) return;
  size_t n = idx / K, k = idx - n * K;
  out[idx] = (bf16)in[k * (size_t)N + n];
}

__global__ void k_feat(const float* __restrict__ z, const float* __restrict__ mu,
                       const float* __restrict__ lv, bf16* __restrict__ feat) {
  size_t idx = (size_t)blockIdx.x * 256 + threadIdx.x;
  if (idx >= (size_t)Bb * Tt * IND) return;
  size_t bt = idx >> 10;
  int d = (int)(idx & 1023);
  float v;
  if (d < 512)      v = z [bt * 512 + d];
  else if (d < 768) v = mu[bt * 256 + (d - 512)];
  else              v = lv[bt * 256 + (d - 768)];
  feat[idx] = (bf16)v;
}

__global__ void k_fill_f32(float* p, float v, size_t n) {
  size_t i = (size_t)blockIdx.x * 256 + threadIdx.x;
  if (i < n) p[i] = v;
}
__global__ void k_fill_b16(bf16* p, size_t n) {
  size_t i = (size_t)blockIdx.x * 256 + threadIdx.x;
  if (i < n) p[i] = (bf16)0.0f;
}

// ---------- generic WMMA GEMM with TDM-staged B -----------------------------
// C[M x N] = A[M x K](bf16,row-major,lda) @ Bt[N x K](bf16,row-major,ldb)^T + bias
// WG = 256 threads (8 waves); per-WG tile 128 x 64; wave w does rows [16w,16w+16).
// B tile (64 rows x KS) staged into LDS by TDM, shared by all 8 waves.
__global__ __launch_bounds__(256) void k_gemm(
    const bf16* __restrict__ A, size_t lda,
    const bf16* __restrict__ Bt, size_t ldb,
    const float* __restrict__ bias,
    float* __restrict__ Cf, bf16* __restrict__ Cb, size_t ldc,
    int K, int relu) {
  __shared__ bf16 sB[64 * KS];   // 16 KB
  const int lane = threadIdx.x & 31;
  const int wave = threadIdx.x >> 5;
  const size_t mbase = (size_t)blockIdx.y * 128 + (size_t)wave * 16;
  const size_t nbase = (size_t)blockIdx.x * 64;

  v8f zero = {0.f, 0.f, 0.f, 0.f, 0.f, 0.f, 0.f, 0.f};
  v8f acc[4] = {zero, zero, zero, zero};

  const bf16* Ap = A + mbase * lda;
  for (int ks = 0; ks < K; ks += KS) {
    if (wave == 0) {
      tdm_load_2d(Bt + nbase * ldb + ks, lds_addr32(sB), KS, 64, (unsigned)ldb);
      __builtin_amdgcn_s_wait_tensorcnt(0);
    }
    __syncthreads();
#pragma unroll
    for (int k2 = 0; k2 < KS; k2 += 32) {
      v16bf a = load_tile(Ap + ks + k2, lda, lane);
#pragma unroll
      for (int ct = 0; ct < 4; ++ct) {
        v16bf b = load_tile(&sB[(ct * 16) * KS + k2], KS, lane);
        acc[ct] = wmma_bf16(a, b, acc[ct]);
      }
    }
    __syncthreads();
  }
  // C/D f32 layout: lane -> N=lane&15; rows base 0 (lanes<16) or 8 (lanes>=16)
  const int n0 = lane & 15;
  const int mr = (lane & 16) ? 8 : 0;
#pragma unroll
  for (int ct = 0; ct < 4; ++ct) {
    size_t ncol = nbase + ct * 16 + n0;
    float bv = bias ? bias[ncol] : 0.0f;
#pragma unroll
    for (int i = 0; i < 8; ++i) {
      float v = acc[ct][i] + bv;
      if (relu) v = fmaxf(v, 0.0f);
      size_t row = mbase + mr + i;
      if (Cf) Cf[row * ldc + ncol] = v;
      if (Cb) Cb[row * ldc + ncol] = (bf16)v;
    }
  }
}

// ---------- fused bidirectional GRU step ------------------------------------
// grid = 32 WGs: blocks 0..15 forward step t_f, blocks 16..31 backward step t_b.
// Each WG owns 64 hidden columns; per K-slice, TDM stages the 3 gate row-blocks
// of the weight matrix into LDS (3 x 64 x KS bf16 = 48 KB), shared by 8 waves.
__global__ __launch_bounds__(256) void k_gru_step(
    const bf16* __restrict__ xall, int t_f, int t_b,
    const bf16* __restrict__ WiT_f, const bf16* __restrict__ WhT_f,
    const float* __restrict__ bi_f, const float* __restrict__ bhn_f,
    const bf16* __restrict__ hpb_f, const float* __restrict__ hpf_f,
    bf16* __restrict__ hnb_f, float* __restrict__ hnf_f,
    const bf16* __restrict__ WiT_b, const bf16* __restrict__ WhT_b,
    const float* __restrict__ bi_b, const float* __restrict__ bhn_b,
    const bf16* __restrict__ hpb_b, const float* __restrict__ hpf_b,
    bf16* __restrict__ hnb_b, float* __restrict__ hnf_b) {
  __shared__ bf16 sB[3 * 64 * KS];   // 48 KB
  const int dir = blockIdx.x >> 4;
  const int wg  = blockIdx.x & 15;
  const int t   = dir ? t_b : t_f;
  const bf16*  WiT = dir ? WiT_b : WiT_f;
  const bf16*  WhT = dir ? WhT_b : WhT_f;
  const float* bi  = dir ? bi_b  : bi_f;
  const float* bhn = dir ? bhn_b : bhn_f;
  const bf16*  hpb = dir ? hpb_b : hpb_f;
  const float* hpf = dir ? hpf_b : hpf_f;
  bf16*  hnb = dir ? hnb_b : hnb_f;
  float* hnf = dir ? hnf_b : hnf_f;

  const int lane = threadIdx.x & 31;
  const int wave = threadIdx.x >> 5;
  const size_t mbase   = (size_t)wave * 16;  // batch rows [mbase, mbase+16)
  const size_t colbase = (size_t)wg * 64;    // hidden cols

  v8f zero = {0.f, 0.f, 0.f, 0.f, 0.f, 0.f, 0.f, 0.f};
  v8f r_acc[4]  = {zero, zero, zero, zero};
  v8f z_acc[4]  = {zero, zero, zero, zero};
  v8f xn_acc[4] = {zero, zero, zero, zero};
  v8f hn_acc[4] = {zero, zero, zero, zero};

  const size_t lda_x = (size_t)Tt * Hh;
  const bf16* xrow0 = xall + (mbase * (size_t)Tt + (size_t)t) * Hh;
  const bf16* hrow0 = hpb + mbase * Hh;

#pragma unroll 1
  for (int phase = 0; phase < 2; ++phase) {
    const bf16* W     = phase ? WhT  : WiT;
    const bf16* Arow0 = phase ? hrow0 : xrow0;
    const size_t ldaA = phase ? (size_t)Hh : lda_x;
#pragma unroll 1
    for (int ks = 0; ks < Hh; ks += KS) {
      if (wave == 0) {
#pragma unroll
        for (int g = 0; g < 3; ++g)
          tdm_load_2d(W + ((size_t)g * Hh + colbase) * Hh + ks,
                      lds_addr32(&sB[g * 64 * KS]), KS, 64, Hh);
        __builtin_amdgcn_s_wait_tensorcnt(0);
      }
      __syncthreads();
#pragma unroll
      for (int k2 = 0; k2 < KS; k2 += 32) {
        v16bf a = load_tile(Arow0 + ks + k2, ldaA, lane);
#pragma unroll
        for (int ct = 0; ct < 4; ++ct) {
          v16bf b0 = load_tile(&sB[(0 * 64 + ct * 16) * KS + k2], KS, lane);
          v16bf b1 = load_tile(&sB[(1 * 64 + ct * 16) * KS + k2], KS, lane);
          v16bf b2 = load_tile(&sB[(2 * 64 + ct * 16) * KS + k2], KS, lane);
          r_acc[ct] = wmma_bf16(a, b0, r_acc[ct]);
          z_acc[ct] = wmma_bf16(a, b1, z_acc[ct]);
          if (phase == 0) xn_acc[ct] = wmma_bf16(a, b2, xn_acc[ct]);
          else            hn_acc[ct] = wmma_bf16(a, b2, hn_acc[ct]);
        }
      }
      __syncthreads();
    }
  }

  // GRU elementwise update
  const int n0 = lane & 15;
  const int mr = (lane & 16) ? 8 : 0;
#pragma unroll
  for (int ct = 0; ct < 4; ++ct) {
    size_t j = colbase + ct * 16 + n0;
    float b_r = bi[j], b_z = bi[Hh + j], b_n = bi[2 * Hh + j], b_hn = bhn[j];
#pragma unroll
    for (int i = 0; i < 8; ++i) {
      size_t row = mbase + mr + i;
      float r  = sigmoidf_(r_acc[ct][i] + b_r);
      float zg = sigmoidf_(z_acc[ct][i] + b_z);
      float nn = tanhf(xn_acc[ct][i] + b_n + r * (hn_acc[ct][i] + b_hn));
      float hp = hpf[row * Hh + j];
      float hv = (1.0f - zg) * nn + zg * hp;
      hnf[row * Hh + j] = hv;
      hnb[row * Hh + j] = (bf16)hv;
    }
  }
}

// ---------- LayerNorm over concat(h_f, h_b) ---------------------------------
__global__ __launch_bounds__(256) void k_layernorm(
    const float* __restrict__ hf, const float* __restrict__ hb,
    const float* __restrict__ g, const float* __restrict__ beta,
    bf16* __restrict__ out) {
  const int row = blockIdx.x;
  const int tid = threadIdx.x;
  const float* h0 = hf + (size_t)row * Hh;
  const float* h1 = hb + (size_t)row * Hh;
  float s = 0.f, s2 = 0.f, vals[8];
#pragma unroll
  for (int j = 0; j < 8; ++j) {
    int idx = tid + j * 256;  // 0..2047
    float v = (idx < Hh) ? h0[idx] : h1[idx - Hh];
    vals[j] = v; s += v; s2 += v * v;
  }
#pragma unroll
  for (int off = 16; off; off >>= 1) {
    s  += __shfl_xor(s,  off, 32);
    s2 += __shfl_xor(s2, off, 32);
  }
  __shared__ float red[16];
  if ((tid & 31) == 0) { red[(tid >> 5) * 2] = s; red[(tid >> 5) * 2 + 1] = s2; }
  __syncthreads();
  if (tid == 0) {
    float a = 0.f, b = 0.f;
    for (int w = 0; w < 8; ++w) { a += red[w * 2]; b += red[w * 2 + 1]; }
    red[0] = a; red[1] = b;
  }
  __syncthreads();
  float mean = red[0] * (1.0f / 2048.0f);
  float var  = red[1] * (1.0f / 2048.0f) - mean * mean;
  float rstd = rsqrtf(var + 1e-6f);
#pragma unroll
  for (int j = 0; j < 8; ++j) {
    int idx = tid + j * 256;
    float v = (vals[j] - mean) * rstd * g[idx] + beta[idx];
    out[(size_t)row * 2048 + idx] = (bf16)v;
  }
}

// ---------- host ------------------------------------------------------------
extern "C" void kernel_launch(void* const* d_in, const int* in_sizes, int n_in,
                              void* d_out, int out_size, void* d_ws, size_t ws_size,
                              hipStream_t stream) {
  (void)in_sizes; (void)n_in; (void)ws_size;
  const float* z      = (const float*)d_in[0];
  const float* mu     = (const float*)d_in[1];
  const float* lv     = (const float*)d_in[2];
  const float* proj_W = (const float*)d_in[3];
  const float* proj_b = (const float*)d_in[4];
  const float* Wi_f   = (const float*)d_in[5];
  const float* Wh_f   = (const float*)d_in[6];
  const float* bi_f   = (const float*)d_in[7];
  const float* bhn_f  = (const float*)d_in[8];
  const float* Wi_b   = (const float*)d_in[9];
  const float* Wh_b   = (const float*)d_in[10];
  const float* bi_b   = (const float*)d_in[11];
  const float* bhn_b  = (const float*)d_in[12];
  const float* ln_g   = (const float*)d_in[13];
  const float* ln_b   = (const float*)d_in[14];
  const float* fc1_W  = (const float*)d_in[15];
  const float* fc1_b  = (const float*)d_in[16];
  const float* fc2_W  = (const float*)d_in[17];
  const float* fc2_b  = (const float*)d_in[18];
  float* out = (float*)d_out;

  char* ws = (char*)d_ws;
  size_t off = 0;
  auto take = [&](size_t bytes) -> char* {
    char* p = ws + off;
    off += (bytes + 255) & ~(size_t)255;
    return p;
  };
  bf16* feat  = (bf16*)take((size_t)Bb * Tt * IND * 2);   // 128 MB
  bf16* xb    = (bf16*)take((size_t)Bb * Tt * Hh * 2);    // 128 MB
  bf16* projT = (bf16*)take((size_t)Hh * IND * 2);
  bf16* WiTf  = (bf16*)take((size_t)3 * Hh * Hh * 2);
  bf16* WhTf  = (bf16*)take((size_t)3 * Hh * Hh * 2);
  bf16* WiTb  = (bf16*)take((size_t)3 * Hh * Hh * 2);
  bf16* WhTb  = (bf16*)take((size_t)3 * Hh * Hh * 2);
  bf16* fc1T  = (bf16*)take((size_t)EMD * 2 * Hh * 2);
  bf16* fc2T  = (bf16*)take((size_t)2 * EMD * EMD * 2);
  float* hf32_f[2]; bf16* hb16_f[2];
  float* hf32_b[2]; bf16* hb16_b[2];
  for (int i = 0; i < 2; ++i) {
    hf32_f[i] = (float*)take((size_t)Bb * Hh * 4);
    hb16_f[i] = (bf16*)take((size_t)Bb * Hh * 2);
    hf32_b[i] = (float*)take((size_t)Bb * Hh * 4);
    hb16_b[i] = (bf16*)take((size_t)Bb * Hh * 2);
  }
  bf16* hln  = (bf16*)take((size_t)Bb * 2 * Hh * 2);
  bf16* fc1o = (bf16*)take((size_t)Bb * EMD * 2);

  auto blocks = [](size_t n) -> unsigned { return (unsigned)((n + 255) / 256); };

  // 1) weight transpose + bf16 convert (kept hot in 192MB L2 thereafter)
  k_transpose_bf16<<<blocks((size_t)Hh * IND), 256, 0, stream>>>(proj_W, projT, IND, Hh);
  k_transpose_bf16<<<blocks((size_t)3 * Hh * Hh), 256, 0, stream>>>(Wi_f, WiTf, Hh, 3 * Hh);
  k_transpose_bf16<<<blocks((size_t)3 * Hh * Hh), 256, 0, stream>>>(Wh_f, WhTf, Hh, 3 * Hh);
  k_transpose_bf16<<<blocks((size_t)3 * Hh * Hh), 256, 0, stream>>>(Wi_b, WiTb, Hh, 3 * Hh);
  k_transpose_bf16<<<blocks((size_t)3 * Hh * Hh), 256, 0, stream>>>(Wh_b, WhTb, Hh, 3 * Hh);
  k_transpose_bf16<<<blocks((size_t)EMD * 2 * Hh), 256, 0, stream>>>(fc1_W, fc1T, 2 * Hh, EMD);
  k_transpose_bf16<<<blocks((size_t)2 * EMD * EMD), 256, 0, stream>>>(fc2_W, fc2T, EMD, 2 * EMD);

  // 2) feat concat -> bf16
  k_feat<<<blocks((size_t)Bb * Tt * IND), 256, 0, stream>>>(z, mu, lv, feat);

  // 3) x = feat @ proj_W + proj_b  (big WMMA GEMM, bf16 out)
  k_gemm<<<dim3(Hh / 64, (Bb * Tt) / 128), 256, 0, stream>>>(
      feat, (size_t)IND, projT, (size_t)IND, proj_b,
      nullptr, xb, (size_t)Hh, IND, 0);

  // 4) zero initial hidden states (buffer 0)
  k_fill_f32<<<blocks((size_t)Bb * Hh), 256, 0, stream>>>(hf32_f[0], 0.0f, (size_t)Bb * Hh);
  k_fill_f32<<<blocks((size_t)Bb * Hh), 256, 0, stream>>>(hf32_b[0], 0.0f, (size_t)Bb * Hh);
  k_fill_b16<<<blocks((size_t)Bb * Hh), 256, 0, stream>>>(hb16_f[0], (size_t)Bb * Hh);
  k_fill_b16<<<blocks((size_t)Bb * Hh), 256, 0, stream>>>(hb16_b[0], (size_t)Bb * Hh);

  // 5) fused bidirectional GRU scan: 512 step kernels (fwd t, bwd T-1-t)
  for (int t = 0; t < Tt; ++t) {
    int cur = t & 1, nxt = (t + 1) & 1;
    k_gru_step<<<32, 256, 0, stream>>>(
        xb, t, Tt - 1 - t,
        WiTf, WhTf, bi_f, bhn_f, hb16_f[cur], hf32_f[cur], hb16_f[nxt], hf32_f[nxt],
        WiTb, WhTb, bi_b, bhn_b, hb16_b[cur], hf32_b[cur], hb16_b[nxt], hf32_b[nxt]);
  }
  // final states land in buffer (Tt & 1) == 0

  // 6) LayerNorm(concat(h_f, h_b)) -> bf16
  k_layernorm<<<Bb, 256, 0, stream>>>(hf32_f[0], hf32_b[0], ln_g, ln_b, hln);

  // 7) fc1 (relu, bf16 out), fc2 (f32 out -> d_out)
  k_gemm<<<dim3(EMD / 64, 1), 256, 0, stream>>>(
      hln, (size_t)(2 * Hh), fc1T, (size_t)(2 * Hh), fc1_b,
      nullptr, fc1o, (size_t)EMD, 2 * Hh, 1);
  k_gemm<<<dim3((2 * EMD) / 64, 1), 256, 0, stream>>>(
      fc1o, (size_t)EMD, fc2T, (size_t)EMD, fc2_b,
      out, nullptr, (size_t)(2 * EMD), EMD, 0);

  // 8) cond_mask = ones (appended after cond in d_out)
  size_t cond_elems = (size_t)Bb * 2 * EMD;
  if ((size_t)out_size > cond_elems) {
    size_t nmask = (size_t)out_size - cond_elems;
    k_fill_f32<<<blocks(nmask), 256, 0, stream>>>(out + cond_elems, 1.0f, nmask);
  }
}
